// Model_85289460564659
// MI455X (gfx1250) — matrix operand order repaired and checked by reading, not compiled, not run on hardware
//
#include <hip/hip_runtime.h>
#include <hip/hip_bf16.h>
#include <math.h>

// ---------------- model constants ----------------
#define T_SEQ   4161      // 1 + 64 + 4096
#define TPRE    65
#define IMG_N   4096
#define DM      768
#define DIN_    1536
#define CONVD   1664      // DIN + 2*64
#define DPROJ_  3224      // 2*DIN + 2*64 + 24
#define NH      24
#define NST     64        // state dim N
#define PDIM    64
#define EPSV    1e-5f

typedef __attribute__((ext_vector_type(16))) _Float16 v16h;
typedef __attribute__((ext_vector_type(2)))  _Float16 half2v;
typedef __attribute__((ext_vector_type(8)))  float    v8f;
typedef __attribute__((ext_vector_type(4)))  int      v4i;

#if __has_builtin(__builtin_amdgcn_global_load_async_to_lds_b128)
#define GEMM_ASYNC 1
// builtin signature (from clang diagnostic): (v4i addrspace(1)*, v4i addrspace(3)*, imm, imm)
#define ASYNC_B128(gsrc, ldst)                                                  \
    __builtin_amdgcn_global_load_async_to_lds_b128(                             \
        (__attribute__((address_space(1))) v4i*)(gsrc),                         \
        (__attribute__((address_space(3))) v4i*)(ldst), 0, 0)
#else
#define GEMM_ASYNC 0
#endif

__device__ __forceinline__ void wait_asynccnt0()
{
#if __has_builtin(__builtin_amdgcn_s_wait_asynccnt)
    __builtin_amdgcn_s_wait_asynccnt(0);
#else
    asm volatile("s_wait_asynccnt 0x0" ::: "memory");
#endif
}

#if GEMM_ASYNC
// Issue async global->LDS copies of one 64x64 f32 A tile and one 64x64 f32 W
// tile (raw, clamped addresses; OOB elements zeroed later in the convert pass).
// Each of 8 iterations: 128 threads x b128 = 2KB per tile per round.
__device__ __forceinline__ void stage_async(
    const float* __restrict__ A, const float* __restrict__ W,
    int M, int K, int Nn, int mode, int wtrans, int dy, int dx,
    int m0, int n0, int k0, int tid, float* As32, float* Bs32)
{
    #pragma unroll
    for (int it = 0; it < 8; it++) {
        int i  = it * 128 + tid;        // b128 chunk id, 1024 per tile
        int r  = i >> 4;
        int c4 = (i & 15) * 4;
        // ---- A tile ----
        {
            const float* src;
            int gm = m0 + r;
            if (mode == 0) {
                int rm = gm < M ? gm : (M - 1);
                src = A + (size_t)rm * K + k0 + c4;
            } else {
                int py = (gm >> 6) + dy, px = (gm & 63) + dx;
                py = py < 0 ? 0 : (py > 63 ? 63 : py);
                px = px < 0 ? 0 : (px > 63 ? 63 : px);
                src = A + ((size_t)(py * 64 + px)) * K + k0 + c4;
            }
            ASYNC_B128(src, &As32[r * 64 + c4]);
        }
        // ---- W tile (wtrans: staged [k][n], else [n][k]) ----
        {
            const float* src;
            if (wtrans) {
                src = W + (size_t)(k0 + r) * Nn + n0 + c4;       // Nn multiple of 64 here
            } else {
                int gn = n0 + r;
                int rn = gn < Nn ? gn : (Nn - 1);
                src = W + (size_t)rn * K + k0 + c4;
            }
            ASYNC_B128(src, &Bs32[r * 64 + c4]);
        }
    }
}
#endif

// =========================================================================
// Generic WMMA GEMM:  C[M,Nn] (+)= A * op(W) (+ bias)
//   A: [M,K] f32 row-major (mode 0) or conv-gather from 64x64xK image (mode 1)
//   W: wtrans==0 -> [Nn,K] row-major (C = A*W^T), wtrans==1 -> [K,Nn] row-major
// Block: 128 threads (4 waves), 64x64 C tile, each wave does a 16x64 strip.
// Async pipeline: global_load_async_to_lds_b128 stages raw f32 tiles; a
// convert pass produces padded f16 tiles; issue of tile k0+64 overlaps the
// v_wmma_f32_16x16x32_f16 section of tile k0.
// =========================================================================
__global__ __launch_bounds__(128)
void gemm_wmma(const float* __restrict__ A, const float* __restrict__ W,
               const float* __restrict__ bias, float* __restrict__ C,
               int M, int K, int Nn, int mode, int accumulate, int wtrans,
               int dy, int dx)
{
    __shared__ _Float16 As[64][72];
    __shared__ _Float16 Bs[64][72];
#if GEMM_ASYNC
    __shared__ float As32[64 * 64];
    __shared__ float Bs32[64 * 64];
#endif

    const int m0   = blockIdx.y * 64;
    const int n0   = blockIdx.x * 64;
    const int tid  = threadIdx.x;
    const int lane = tid & 31;
    const int wv   = tid >> 5;          // wave id 0..3 -> M sub-tile
    const int lid  = lane & 15;
    const int hi   = lane >> 4;         // lane 16..31 flag

    v8f zero = {};
    v8f acc0 = zero, acc1 = zero, acc2 = zero, acc3 = zero;

#if GEMM_ASYNC
    stage_async(A, W, M, K, Nn, mode, wtrans, dy, dx, m0, n0, 0, tid, As32, Bs32);
#endif

    for (int k0 = 0; k0 < K; k0 += 64) {
#if GEMM_ASYNC
        // ---- wait for staged f32 tiles, convert to padded f16 tiles ----
        wait_asynccnt0();
        __syncthreads();
        for (int i = tid; i < 64 * 64; i += 128) {
            int r = i >> 6, c = i & 63;
            int gm = m0 + r;
            bool va = (gm < M);
            if (mode == 1) {
                int py = (gm >> 6) + dy, px = (gm & 63) + dx;
                va = va && (py >= 0) && (py < 64) && (px >= 0) && (px < 64);
            }
            As[r][c] = (_Float16)(va ? As32[i] : 0.f);
        }
        for (int i = tid; i < 64 * 64; i += 128) {
            int r, c;
            if (wtrans) { r = i & 63; c = i >> 6; }
            else        { r = i >> 6; c = i & 63; }
            float bv = wtrans ? Bs32[c * 64 + r] : Bs32[r * 64 + c];
            bool vb = wtrans ? true : ((n0 + r) < Nn);
            Bs[r][c] = (_Float16)(vb ? bv : 0.f);
        }
        __syncthreads();
        // ---- overlap: issue async copies of the NEXT k-tile now ----
        if (k0 + 64 < K)
            stage_async(A, W, M, K, Nn, mode, wtrans, dy, dx,
                        m0, n0, k0 + 64, tid, As32, Bs32);
#else
        // ---- fallback: synchronous staged conversion ----
        for (int i = tid; i < 64 * 64; i += 128) {
            int r = i >> 6, c = i & 63;
            int gm = m0 + r, gk = k0 + c;
            float v = 0.f;
            if (gk < K) {
                if (mode == 0) {
                    if (gm < M) v = A[(size_t)gm * K + gk];
                } else {
                    int py = (gm >> 6) + dy, px = (gm & 63) + dx;
                    if (gm < M && py >= 0 && py < 64 && px >= 0 && px < 64)
                        v = A[((size_t)(py * 64 + px)) * K + gk];
                }
            }
            As[r][c] = (_Float16)v;
        }
        for (int i = tid; i < 64 * 64; i += 128) {
            int r = i >> 6, c = i & 63;
            int gn = n0 + r, gk = k0 + c;
            float v = 0.f;
            if (gn < Nn && gk < K)
                v = wtrans ? W[(size_t)gk * Nn + gn] : W[(size_t)gn * K + gk];
            Bs[r][c] = (_Float16)v;
        }
        __syncthreads();
        if (mode == 0 && k0 + 64 < K) {
            int gm = m0 + (tid >> 1);
            if (gm < M)
                __builtin_prefetch(&A[(size_t)gm * K + k0 + 64 + (tid & 1) * 32], 0, 1);
        }
#endif

        #pragma unroll
        for (int kk = 0; kk < 64; kk += 32) {
            // A fragment: row = wave's 16-row strip, K-pairs per ISA layout
            v16h afr;
            {
                const half2v* ap = (const half2v*)(&As[wv * 16 + lid][0]);
                #pragma unroll
                for (int j = 0; j < 8; j++) {
                    int kb = ((j < 4) ? (2 * j) : (16 + 2 * (j - 4))) + (hi ? 8 : 0);
                    half2v pr = ap[(kk + kb) >> 1];
                    afr[2 * j] = pr[0];
                    afr[2 * j + 1] = pr[1];
                }
            }
            #pragma unroll
            for (int nt = 0; nt < 4; nt++) {
                v16h bfr;
                const half2v* bp = (const half2v*)(&Bs[nt * 16 + lid][0]);
                #pragma unroll
                for (int j = 0; j < 8; j++) {
                    int kb = ((j < 4) ? (2 * j) : (16 + 2 * (j - 4))) + (hi ? 8 : 0);
                    half2v pr = bp[(kk + kb) >> 1];
                    bfr[2 * j] = pr[0];
                    bfr[2 * j + 1] = pr[1];
                }
                v8f* accp = (nt == 0) ? &acc0 : (nt == 1) ? &acc1 : (nt == 2) ? &acc2 : &acc3;
                *accp = __builtin_amdgcn_wmma_f32_16x16x32_f16(
                    false, afr, false, bfr, (short)0, *accp, false, false);
            }
        }
        __syncthreads();
    }

    // ---- epilogue: C layout per ISA (lane -> col, vgpr v -> row v (+8 for hi)) ----
    #pragma unroll
    for (int nt = 0; nt < 4; nt++) {
        int gn = n0 + nt * 16 + lid;
        if (gn >= Nn) continue;
        float bv = bias ? bias[gn] : 0.f;
        v8f accv = (nt == 0) ? acc0 : (nt == 1) ? acc1 : (nt == 2) ? acc2 : acc3;
        #pragma unroll
        for (int v = 0; v < 8; v++) {
            int gm = m0 + wv * 16 + v + (hi ? 8 : 0);
            if (gm < M) {
                float val = accv[v] + bv;
                size_t off = (size_t)gm * Nn + gn;
                if (accumulate) C[off] += val;
                else            C[off] = val;
            }
        }
    }
}

// =========================================================================
// Elementwise / scan kernels
// =========================================================================
__device__ __forceinline__ float siluf(float x) { return x / (1.f + __expf(-x)); }

__global__ void k_fromrgb(const float* __restrict__ im8, const float* __restrict__ w,
                          const float* __restrict__ b, float* __restrict__ e8)
{
    int idx = blockIdx.x * blockDim.x + threadIdx.x;     // 64*768
    if (idx >= 64 * DM) return;
    int i = idx / DM, d = idx % DM;
    float v = im8[i * 3 + 0] * w[d * 3 + 0] + im8[i * 3 + 1] * w[d * 3 + 1] +
              im8[i * 3 + 2] * w[d * 3 + 2] + b[d];
    e8[idx] = v;
}

__global__ void k_build_x(const float* __restrict__ e8, const float* __restrict__ s0,
                          const float* __restrict__ suffix, float* __restrict__ x)
{
    int t = blockIdx.x;
    #pragma unroll
    for (int i = 0; i < 3; i++) {
        int d = threadIdx.x + 256 * i;
        float v;
        if (t == 0)             v = s0[d];
        else if (t < TPRE)      v = e8[(t - 1) * DM + d];
        else {
            int p = t - TPRE;
            int py = p >> 6, px = p & 63;
            int e = (py >> 3) * 8 + (px >> 3);
            v = e8[e * DM + d] + suffix[(size_t)p * DM + d];
        }
        x[(size_t)t * DM + d] = v;
    }
}

__global__ void k_layernorm(const float* __restrict__ src, float* __restrict__ dst,
                            const float* __restrict__ w, const float* __restrict__ b,
                            int rows)
{
    int t = blockIdx.x;
    if (t >= rows) return;
    __shared__ float r1[256], r2[256];
    float v[3]; float s = 0.f, sq = 0.f;
    #pragma unroll
    for (int i = 0; i < 3; i++) {
        int d = threadIdx.x + 256 * i;
        float x = src[(size_t)t * DM + d];
        v[i] = x; s += x; sq += x * x;
    }
    r1[threadIdx.x] = s; r2[threadIdx.x] = sq;
    __syncthreads();
    for (int st = 128; st > 0; st >>= 1) {
        if (threadIdx.x < st) { r1[threadIdx.x] += r1[threadIdx.x + st];
                                r2[threadIdx.x] += r2[threadIdx.x + st]; }
        __syncthreads();
    }
    float mean = r1[0] * (1.f / DM);
    float var  = r2[0] * (1.f / DM) - mean * mean;
    float inv  = rsqrtf(var + EPSV);
    #pragma unroll
    for (int i = 0; i < 3; i++) {
        int d = threadIdx.x + 256 * i;
        dst[(size_t)t * DM + d] = (v[i] - mean) * inv * w[d] + b[d];
    }
}

__global__ void k_dt(const float* __restrict__ zx, const float* __restrict__ dtb,
                     float* __restrict__ dt)
{
    int idx = blockIdx.x * blockDim.x + threadIdx.x;
    if (idx >= T_SEQ * NH) return;
    int t = idx / NH, h = idx % NH;
    float v = zx[(size_t)t * DPROJ_ + (DPROJ_ - NH) + h] + dtb[h];
    dt[idx] = (v > 20.f) ? v : log1pf(__expf(v));
}

__global__ void k_conv1d(const float* __restrict__ zx, const float* __restrict__ cw,
                         const float* __restrict__ cb, float* __restrict__ xbc)
{
    int idx = blockIdx.x * blockDim.x + threadIdx.x;
    if (idx >= T_SEQ * CONVD) return;
    int t = idx / CONVD, c = idx % CONVD;
    float acc = cb[c];
    #pragma unroll
    for (int j = 0; j < 4; j++) {
        int tt = t - 3 + j;
        if (tt >= 0) acc += zx[(size_t)tt * DPROJ_ + DIN_ + c] * cw[c * 4 + j];
    }
    xbc[idx] = siluf(acc);
}

__global__ void k_yinit(const float* __restrict__ xbc, const float* __restrict__ Dp,
                        float* __restrict__ y)
{
    int idx = blockIdx.x * blockDim.x + threadIdx.x;
    if (idx >= T_SEQ * DIN_) return;
    int t = idx / DIN_, d = idx % DIN_;
    y[idx] = Dp[d >> 6] * xbc[(size_t)t * CONVD + d];
}

// Sequential SSM scan; grid (24 heads, 4 state-chunks), 64 threads = p dim.
// Each thread carries 16 state entries in registers; partial y via atomicAdd.
__global__ __launch_bounds__(64)
void k_ssd_scan(const float* __restrict__ xbc, const float* __restrict__ dt,
                const float* __restrict__ A_log, float* __restrict__ y)
{
    int h  = blockIdx.x;
    int nb = blockIdx.y * 16;
    int p  = threadIdx.x;
    float Aval = -__expf(A_log[h]);
    float S[16];
    #pragma unroll
    for (int i = 0; i < 16; i++) S[i] = 0.f;

    for (int t = 0; t < T_SEQ; t++) {
        const float* row = xbc + (size_t)t * CONVD;
        float d  = dt[t * NH + h];
        float dA = __expf(d * Aval);
        float xp = row[h * PDIM + p] * d;
        float accv = 0.f;
        #pragma unroll
        for (int i = 0; i < 16; i++) {
            float Bv = row[DIN_ + nb + i];
            float Cv = row[DIN_ + NST + nb + i];
            S[i] = S[i] * dA + xp * Bv;
            accv += Cv * S[i];
        }
        atomicAdd(&y[(size_t)t * DIN_ + h * PDIM + p], accv);
    }
}

__global__ void k_gate_rms(float* __restrict__ y, const float* __restrict__ zx,
                           const float* __restrict__ rmsw)
{
    int t = blockIdx.x;
    __shared__ float red[256];
    float v[6]; float ss = 0.f;
    #pragma unroll
    for (int i = 0; i < 6; i++) {
        int d = threadIdx.x + 256 * i;
        float z = zx[(size_t)t * DPROJ_ + d];
        float val = y[(size_t)t * DIN_ + d] * siluf(z);
        v[i] = val; ss += val * val;
    }
    red[threadIdx.x] = ss;
    __syncthreads();
    for (int st = 128; st > 0; st >>= 1) {
        if (threadIdx.x < st) red[threadIdx.x] += red[threadIdx.x + st];
        __syncthreads();
    }
    float scale = rsqrtf(red[0] * (1.f / DIN_) + EPSV);
    #pragma unroll
    for (int i = 0; i < 6; i++) {
        int d = threadIdx.x + 256 * i;
        y[(size_t)t * DIN_ + d] = v[i] * scale * rmsw[d];
    }
}

// x_next = swap(x + h), h image part replaced by conv output
__global__ void k_residual_swap(const float* __restrict__ x, const float* __restrict__ h2,
                                const float* __restrict__ cv, float* __restrict__ xn)
{
    int t = blockIdx.x;
    #pragma unroll
    for (int i = 0; i < 3; i++) {
        int d = threadIdx.x + 256 * i;
        float v;
        if (t < TPRE) {
            v = x[(size_t)t * DM + d] + h2[(size_t)t * DM + d];
        } else {
            int o = t - TPRE;
            int src = (IMG_N - 1) - o;     // flip
            v = x[(size_t)(TPRE + src) * DM + d] + cv[(size_t)src * DM + d];
        }
        xn[(size_t)t * DM + d] = v;
    }
}

__global__ void k_zero_loss(float* out) { out[IMG_N * 3] = 0.f; }

__global__ void k_head(const float* __restrict__ ximg, const float* __restrict__ w,
                       const float* __restrict__ b, const float* __restrict__ im64,
                       float* __restrict__ out)
{
    int p = blockIdx.x;
    __shared__ float red[3][256];
    float a0 = 0.f, a1 = 0.f, a2 = 0.f;
    #pragma unroll
    for (int i = 0; i < 3; i++) {
        int d = threadIdx.x + 256 * i;
        float xv = ximg[(size_t)p * DM + d];
        a0 += xv * w[0 * DM + d];
        a1 += xv * w[1 * DM + d];
        a2 += xv * w[2 * DM + d];
    }
    red[0][threadIdx.x] = a0; red[1][threadIdx.x] = a1; red[2][threadIdx.x] = a2;
    __syncthreads();
    for (int st = 128; st > 0; st >>= 1) {
        if (threadIdx.x < st) {
            red[0][threadIdx.x] += red[0][threadIdx.x + st];
            red[1][threadIdx.x] += red[1][threadIdx.x + st];
            red[2][threadIdx.x] += red[2][threadIdx.x + st];
        }
        __syncthreads();
    }
    if (threadIdx.x == 0) {
        float lpart = 0.f;
        #pragma unroll
        for (int c = 0; c < 3; c++) {
            float yv = red[c][0] + b[c];
            out[p * 3 + c] = yv;
            float diff = yv - im64[p * 3 + c];
            lpart += diff * diff;
        }
        atomicAdd(&out[IMG_N * 3], lpart * (1.f / (IMG_N * 3)));
    }
}

// =========================================================================
// Host orchestration
// =========================================================================
static inline void launch_gemm(const float* A, const float* W, const float* bias,
                               float* C, int M, int K, int Nn, int mode, int acc,
                               int wtrans, int dy, int dx, hipStream_t s)
{
    dim3 g((Nn + 63) / 64, (M + 63) / 64);
    gemm_wmma<<<g, 128, 0, s>>>(A, W, bias, C, M, K, Nn, mode, acc, wtrans, dy, dx);
}

extern "C" void kernel_launch(void* const* d_in, const int* in_sizes, int n_in,
                              void* d_out, int out_size, void* d_ws, size_t ws_size,
                              hipStream_t stream)
{
    const float* im8        = (const float*)d_in[0];
    const float* im64       = (const float*)d_in[1];
    const float* from_rgb_w = (const float*)d_in[2];
    const float* from_rgb_b = (const float*)d_in[3];
    const float* to_rgb_w   = (const float*)d_in[4];
    const float* to_rgb_b   = (const float*)d_in[5];
    const float* s0         = (const float*)d_in[6];
    const float* suffix     = (const float*)d_in[7];
    const float* norm0_w    = (const float*)d_in[8];
    const float* norm0_b    = (const float*)d_in[9];
    const float* in_proj_w  = (const float*)d_in[10];
    const float* conv1d_w   = (const float*)d_in[11];
    const float* conv1d_b   = (const float*)d_in[12];
    const float* dt_bias    = (const float*)d_in[13];
    const float* A_log      = (const float*)d_in[14];
    const float* Dp         = (const float*)d_in[15];
    const float* rms_w      = (const float*)d_in[16];
    const float* out_proj_w = (const float*)d_in[17];
    const float* ln_w       = (const float*)d_in[18];
    const float* ln_b       = (const float*)d_in[19];
    const float* lnc_w      = (const float*)d_in[20];
    const float* lnc_b      = (const float*)d_in[21];
    const float* conv2d_w   = (const float*)d_in[22];
    const float* conv2d_b   = (const float*)d_in[23];

    float* ws = (float*)d_ws;
    size_t off = 0;
    auto alloc = [&](size_t n) { float* p = ws + off; off += n; return p; };
    float* e8   = alloc((size_t)64 * DM);
    float* xA   = alloc((size_t)T_SEQ * DM);
    float* xB   = alloc((size_t)T_SEQ * DM);
    float* hbuf = alloc((size_t)T_SEQ * DM);
    float* zx   = alloc((size_t)T_SEQ * DPROJ_);
    float* xbc  = alloc((size_t)T_SEQ * CONVD);
    float* dtb  = alloc((size_t)T_SEQ * NH);
    float* ybuf = alloc((size_t)T_SEQ * DIN_);    // reused for imgln + convout after out_proj
    float* h2   = alloc((size_t)T_SEQ * DM);
    float* imgln = ybuf;                          // IMG_N*DM <= half of ybuf
    float* cvout = ybuf + (size_t)IMG_N * DM;

    float* x  = xA;
    float* xn = xB;

    // ---- embedding + norm0 ----
    k_fromrgb<<<(64 * DM + 255) / 256, 256, 0, stream>>>(im8, from_rgb_w, from_rgb_b, e8);
    k_build_x<<<T_SEQ, 256, 0, stream>>>(e8, s0, suffix, x);
    k_layernorm<<<T_SEQ, 256, 0, stream>>>(x, x, norm0_w, norm0_b, T_SEQ);

    for (int l = 0; l < 8; l++) {
        const float* inw  = in_proj_w  + (size_t)l * DPROJ_ * DM;
        const float* outw = out_proj_w + (size_t)l * DM * DIN_;

        k_layernorm<<<T_SEQ, 256, 0, stream>>>(x, hbuf, ln_w + l * DM, ln_b + l * DM, T_SEQ);

        // in_proj: zx[T,3224] = h * inw^T   (WMMA)
        launch_gemm(hbuf, inw, nullptr, zx, T_SEQ, DM, DPROJ_, 0, 0, 0, 0, 0, stream);

        k_dt<<<(T_SEQ * NH + 255) / 256, 256, 0, stream>>>(zx, dt_bias + l * NH, dtb);
        k_conv1d<<<(T_SEQ * CONVD + 255) / 256, 256, 0, stream>>>(
            zx, conv1d_w + (size_t)l * CONVD * 4, conv1d_b + (size_t)l * CONVD, xbc);

        k_yinit<<<(T_SEQ * DIN_ + 255) / 256, 256, 0, stream>>>(xbc, Dp + l * NH, ybuf);
        k_ssd_scan<<<dim3(NH, 4), 64, 0, stream>>>(xbc, dtb, A_log + l * NH, ybuf);

        k_gate_rms<<<T_SEQ, 256, 0, stream>>>(ybuf, zx, rms_w + (size_t)l * DIN_);

        // out_proj: h2[T,768] = y * outw^T  (WMMA)
        launch_gemm(ybuf, outw, nullptr, h2, T_SEQ, DIN_, DM, 0, 0, 0, 0, 0, stream);

        // lnc on image part, then 5x5 conv as 25 accumulating WMMA tap-GEMMs
        k_layernorm<<<IMG_N, 256, 0, stream>>>(h2 + (size_t)TPRE * DM, imgln,
                                               lnc_w + l * DM, lnc_b + l * DM, IMG_N);
        for (int ky = 0; ky < 5; ky++) {
            for (int kx = 0; kx < 5; kx++) {
                const float* wt = conv2d_w +
                    ((size_t)l * 25 + (size_t)(ky * 5 + kx)) * (size_t)DM * DM;
                int first = (ky == 0 && kx == 0);
                launch_gemm(imgln, wt, first ? (conv2d_b + l * DM) : nullptr, cvout,
                            IMG_N, DM, DM, /*mode=*/1, /*acc=*/first ? 0 : 1,
                            /*wtrans=*/1, ky - 2, kx - 2, stream);
            }
        }

        k_residual_swap<<<T_SEQ, 256, 0, stream>>>(x, h2, cvout, xn);
        float* tmp = x; x = xn; xn = tmp;
    }

    // ---- head: y_hat + mse loss ----
    k_zero_loss<<<1, 1, 0, stream>>>((float*)d_out);
    k_head<<<IMG_N, 256, 0, stream>>>(x + (size_t)TPRE * DM, to_rgb_w, to_rgb_b,
                                      im64, (float*)d_out);
}
